// GNNFusionModel_27230092657435
// MI455X (gfx1250) — matrix operand order repaired and checked by reading, not compiled, not run on hardware
//
#include <hip/hip_runtime.h>
#include <hip/hip_bf16.h>

// ---------------------------------------------------------------------------
// Types for CDNA5 WMMA (wave32, 16x16x32 bf16 -> f32) and TDM descriptors
// ---------------------------------------------------------------------------
typedef __attribute__((ext_vector_type(16))) __bf16 bf16x16;
typedef __attribute__((ext_vector_type(8)))  __bf16 bf16x8;
typedef __attribute__((ext_vector_type(8)))  float  f32x8;
typedef __attribute__((ext_vector_type(4)))  unsigned int tdm_u32x4;
typedef __attribute__((ext_vector_type(8)))  int          tdm_i32x8;
typedef __attribute__((ext_vector_type(4)))  int          tdm_i32x4;

#define NNODES 10000
#define SEQ    16
#define DMODEL 768
#define KPAD   800      // 769 padded to multiple of 32
#define F1     512
#define F2     256
#define F3     64

__device__ __forceinline__ f32x8 wmma_bf16(bf16x16 a, bf16x16 b, f32x8 c) {
  return __builtin_amdgcn_wmma_f32_16x16x32_bf16(
      /*neg_a=*/false, a, /*neg_b=*/false, b,
      /*c_mod=*/(short)0, c, /*reuse_a=*/false, /*reuse_b=*/false);
}

// A fragment (16x32 bf16, row-major source, row stride lda elements).
// Lane m = lane&15; K contiguous: 0..15 (low) / 16..31 (high).
__device__ __forceinline__ bf16x16 frag_a(const __bf16* tile, int lda, int lane) {
  const int m  = lane & 15;
  const int ko = (lane & 16) ? 16 : 0;
  const __bf16* p = tile + (size_t)m * lda + ko;
  bf16x8 lo = *reinterpret_cast<const bf16x8*>(p);
  bf16x8 hi = *reinterpret_cast<const bf16x8*>(p + 8);
  bf16x16 r;
#pragma unroll
  for (int i = 0; i < 8; ++i) { r[i] = lo[i]; r[i + 8] = hi[i]; }
  return r;
}

// B fragment (32x16 bf16) from a torch-layout weight Wt[n][k] (row-major,
// row stride ldb). Lane n = lane&15; K contiguous: 0..15 (low) / 16..31 (high).
__device__ __forceinline__ bf16x16 frag_b(const __bf16* tile, int ldb, int lane) {
  const int n  = lane & 15;
  const int ko = (lane & 16) ? 16 : 0;
  const __bf16* p = tile + (size_t)n * ldb + ko;
  bf16x8 lo = *reinterpret_cast<const bf16x8*>(p);
  bf16x8 hi = *reinterpret_cast<const bf16x8*>(p + 8);
  bf16x16 r;
#pragma unroll
  for (int i = 0; i < 8; ++i) { r[i] = lo[i]; r[i + 8] = hi[i]; }
  return r;
}

// ---------------------------------------------------------------------------
// Prep: f32 -> bf16 cast (torch-layout weights keep layout)
// ---------------------------------------------------------------------------
__global__ void cvt_bf16_kernel(const float* __restrict__ src,
                                __bf16* __restrict__ dst, int n) {
  int i = blockIdx.x * blockDim.x + threadIdx.x;
  if (i < n) dst[i] = (__bf16)src[i];
}

// Transpose [K,Nout] (x@W layout) -> bf16 [Nout,Kpad] with zero K-padding.
__global__ void transpose_cvt_kernel(const float* __restrict__ W,
                                     __bf16* __restrict__ Wt,
                                     int K, int Nout, int Kpad) {
  int i = blockIdx.x * blockDim.x + threadIdx.x;
  if (i >= Nout * Kpad) return;
  int o = i / Kpad, k = i % Kpad;
  Wt[i] = (__bf16)(k < K ? W[(size_t)k * Nout + o] : 0.0f);
}

// ---------------------------------------------------------------------------
// Fused attention block: one workgroup (8 waves) per node.
//   TDM-stage x[16,768] -> qkv -> softmax(q k^T/sqrt(D)) -> o=attn v
//   -> out_proj -> mean over S -> layernorm -> nf[node] = [scalar|fv|pad]
// ---------------------------------------------------------------------------
__global__ __launch_bounds__(256) void attn_kernel(
    const float* __restrict__ fv, const float* __restrict__ scalar_in,
    const __bf16* __restrict__ wq, const float* __restrict__ in_b,
    const __bf16* __restrict__ wo, const float* __restrict__ out_b,
    const float* __restrict__ ln_g, const float* __restrict__ ln_b,
    __bf16* __restrict__ nf) {
  extern __shared__ char smem_raw[];
  __bf16* xs     = (__bf16*)smem_raw;          // 16*768  (x bf16, later reused for o)
  __bf16* qkvs   = xs + 16 * DMODEL;           // 16*2304
  __bf16* at     = qkvs + 16 * 3 * DMODEL;     // 16*32 (attn, K-padded w/ zeros)
  float*  pooled = (float*)(at + 16 * 32);     // 768
  float*  red    = pooled + DMODEL;            // 512
  float*  xf32   = (float*)qkvs;               // 16*768 f32 TDM staging (aliases qkv)

  const int node = blockIdx.x;
  const int tid  = threadIdx.x;
  const int lane = tid & 31;
  const int wave = tid >> 5;
  const int n16  = lane & 15;
  const int mb   = (lane & 16) ? 8 : 0;

  // ---- TDM: DMA the contiguous 16x768 f32 tile into LDS (wave 0 issues) ----
  if (wave == 0) {
    const unsigned long long ga =
        (unsigned long long)(const void*)(fv + (size_t)node * (SEQ * DMODEL));
    const unsigned ldsoff = (unsigned)(unsigned long long)(void*)xf32;
    const unsigned ne = SEQ * DMODEL;          // 12288 elements, one "row"
    tdm_u32x4 g0;
    g0[0] = 1u;                                // count=1, user descriptor
    g0[1] = ldsoff;                            // lds_addr (bytes)
    g0[2] = (unsigned)(ga & 0xffffffffu);      // global_addr[31:0]
    g0[3] = (unsigned)(ga >> 32) | 0x80000000u;// global_addr[56:32] | type=2
    tdm_i32x8 g1;
    g1[0] = (int)(2u << 16);                   // workgroup_mask=0, data_size=4B
    g1[1] = (int)((ne & 0xffffu) << 16);       // tensor_dim0[15:0]
    g1[2] = (int)((ne >> 16) | (1u << 16));    // tensor_dim0[31:16] | tensor_dim1=1
    g1[3] = (int)((ne & 0xffffu) << 16);       // tile_dim0 = 12288
    g1[4] = 1;                                 // tile_dim1 = 1 (tile_dim2 = 0)
    g1[5] = (int)ne;                           // tensor_dim0_stride[31:0]
    g1[6] = 0;                                 // stride0 hi | stride1 lo
    g1[7] = 0;
    tdm_i32x4 z4 = {};
    tdm_i32x8 z8 = {};
    __builtin_amdgcn_tensor_load_to_lds(g0, g1, z4, z4, z8, 0);
    __builtin_amdgcn_s_wait_tensorcnt(0);
  }
  __syncthreads();
  // convert staged f32 -> bf16 (xf32 aliases qkv region; xs is disjoint)
  for (int i = tid; i < SEQ * DMODEL; i += 256) xs[i] = (__bf16)xf32[i];
  __syncthreads();

  // ---- QKV: 144 n-tiles = 8 waves * 9 pairs; K = 768 (24 steps) ----
  for (int t = 0; t < 9; ++t) {
    const int nt0 = wave * 18 + t * 2;
    f32x8 acc0 = {}, acc1 = {};
#pragma unroll 4
    for (int kt = 0; kt < 24; ++kt) {
      bf16x16 a  = frag_a(xs + kt * 32, DMODEL, lane);
      bf16x16 b0 = frag_b(wq + (size_t)(nt0 * 16) * DMODEL + kt * 32, DMODEL, lane);
      bf16x16 b1 = frag_b(wq + (size_t)((nt0 + 1) * 16) * DMODEL + kt * 32, DMODEL, lane);
      acc0 = wmma_bf16(a, b0, acc0);
      acc1 = wmma_bf16(a, b1, acc1);
    }
    const int n0 = nt0 * 16 + n16;
    const float bi0 = in_b[n0], bi1 = in_b[n0 + 16];
#pragma unroll
    for (int r = 0; r < 8; ++r) {
      qkvs[(mb + r) * (3 * DMODEL) + n0]      = (__bf16)(acc0[r] + bi0);
      qkvs[(mb + r) * (3 * DMODEL) + n0 + 16] = (__bf16)(acc1[r] + bi1);
    }
  }
  __syncthreads();

  // ---- scores = q k^T / sqrt(D), softmax (wave 0 only) ----
  if (wave == 0) {
    f32x8 acc = {};
#pragma unroll 4
    for (int kt = 0; kt < 24; ++kt) {
      bf16x16 a = frag_a(qkvs + kt * 32, 3 * DMODEL, lane);          // q
      bf16x16 b = frag_b(qkvs + DMODEL + kt * 32, 3 * DMODEL, lane); // k (col n = seq s)
      acc = wmma_bf16(a, b, acc);
    }
    const float sc = 0.03608439182435161f; // 1/sqrt(768)
#pragma unroll
    for (int r = 0; r < 8; ++r) {
      float v = acc[r] * sc;
      float m = v;
      for (int d = 1; d < 16; d <<= 1) m = fmaxf(m, __shfl_xor(m, d, 32));
      float e = __expf(v - m);
      float s = e;
      for (int d = 1; d < 16; d <<= 1) s += __shfl_xor(s, d, 32);
      at[(mb + r) * 32 + n16]      = (__bf16)(e / s);
      at[(mb + r) * 32 + 16 + n16] = (__bf16)0.0f;   // zero K-pad 16..31
    }
  }
  __syncthreads();

  // ---- o = attn @ v : 48 n-tiles (8 waves * 6), single K=32 step ----
  const bf16x16 a_at = frag_a(at, 32, lane);
  for (int t = 0; t < 6; ++t) {
    const int nt   = wave * 6 + t;
    const int dcol = nt * 16 + n16;
    bf16x16 b;
    if (lane < 16) {
#pragma unroll
      for (int s = 0; s < 16; ++s) b[s] = qkvs[s * (3 * DMODEL) + 2 * DMODEL + dcol];
    } else {
#pragma unroll
      for (int s = 0; s < 16; ++s) b[s] = (__bf16)0.0f;   // K 16..31 pad
    }
    f32x8 acc = {};
    acc = wmma_bf16(a_at, b, acc);
#pragma unroll
    for (int r = 0; r < 8; ++r) xs[(mb + r) * DMODEL + dcol] = (__bf16)acc[r]; // xs := o
  }
  __syncthreads();

  // ---- y = o @ Wout^T + b, fused mean-pool over S (3 pairs per wave) ----
  for (int t = 0; t < 3; ++t) {
    const int nt0 = wave * 6 + t * 2;
    f32x8 acc0 = {}, acc1 = {};
#pragma unroll 4
    for (int kt = 0; kt < 24; ++kt) {
      bf16x16 a  = frag_a(xs + kt * 32, DMODEL, lane);
      bf16x16 b0 = frag_b(wo + (size_t)(nt0 * 16) * DMODEL + kt * 32, DMODEL, lane);
      bf16x16 b1 = frag_b(wo + (size_t)((nt0 + 1) * 16) * DMODEL + kt * 32, DMODEL, lane);
      acc0 = wmma_bf16(a, b0, acc0);
      acc1 = wmma_bf16(a, b1, acc1);
    }
    const int n0 = nt0 * 16 + n16;
    float cs0 = 0.0f, cs1 = 0.0f;
#pragma unroll
    for (int r = 0; r < 8; ++r) { cs0 += acc0[r]; cs1 += acc1[r]; }
    cs0 += __shfl_xor(cs0, 16, 32);           // combine row-halves
    cs1 += __shfl_xor(cs1, 16, 32);
    if (lane < 16) {
      pooled[n0]      = cs0 * (1.0f / 16.0f) + out_b[n0];
      pooled[n0 + 16] = cs1 * (1.0f / 16.0f) + out_b[n0 + 16];
    }
  }
  __syncthreads();

  // ---- layernorm over 768, emit node feature row (bf16, width KPAD) ----
  float s1 = 0.0f, s2 = 0.0f;
  for (int i = tid; i < DMODEL; i += 256) { float v = pooled[i]; s1 += v; s2 += v * v; }
  red[tid] = s1; red[256 + tid] = s2;
  __syncthreads();
  for (int off = 128; off > 0; off >>= 1) {
    if (tid < off) { red[tid] += red[tid + off]; red[256 + tid] += red[256 + tid + off]; }
    __syncthreads();
  }
  const float mu  = red[0] * (1.0f / DMODEL);
  const float var = red[256] * (1.0f / DMODEL) - mu * mu;
  const float rs  = rsqrtf(var + 1e-5f);
  __bf16* nfr = nf + (size_t)node * KPAD;
  if (tid == 0) nfr[0] = (__bf16)scalar_in[node];
  for (int i = tid; i < DMODEL; i += 256)
    nfr[1 + i] = (__bf16)((pooled[i] - mu) * rs * ln_g[i] + ln_b[i]);
  if (tid < KPAD - DMODEL - 1) nfr[DMODEL + 1 + tid] = (__bf16)0.0f;
}

// ---------------------------------------------------------------------------
// Generic bf16 WMMA GEMM, 1x2 register blocking: each wave computes a 16x32
// C slab, reusing every A fragment for two WMMAs (two independent XDL chains).
// A row-major bf16 (lda), Bt torch-layout bf16 [N,K] (ldb).
// ---------------------------------------------------------------------------
template <bool RELU, bool HAS_BIAS>
__global__ __launch_bounds__(128) void gemm_kernel(
    const __bf16* __restrict__ A, int lda,
    const __bf16* __restrict__ Bt, int ldb,
    const float* __restrict__ bias,
    float* __restrict__ C, int ldc,
    int Ntiles, int Ksteps) {
  const int lane = threadIdx.x & 31;
  const int wave = threadIdx.x >> 5;
  const int nt0  = (blockIdx.y * 4 + wave) * 2;
  if (nt0 >= Ntiles) return;
  const bool has2 = (nt0 + 1) < Ntiles;
  const size_t m0 = (size_t)blockIdx.x * 16;
  const __bf16* Ab = A + m0 * lda;
  const __bf16* B0 = Bt + (size_t)nt0 * 16 * ldb;
  const __bf16* B1 = B0 + (size_t)16 * ldb;
  f32x8 acc0 = {}, acc1 = {};
#pragma unroll 2
  for (int kt = 0; kt < Ksteps; ++kt) {
    __builtin_prefetch(Ab + (kt + 4) * 32, 0, 0);   // global_prefetch_b8
    bf16x16 a  = frag_a(Ab + kt * 32, lda, lane);
    bf16x16 b0 = frag_b(B0 + kt * 32, ldb, lane);
    acc0 = wmma_bf16(a, b0, acc0);
    if (has2) {
      bf16x16 b1 = frag_b(B1 + kt * 32, ldb, lane);
      acc1 = wmma_bf16(a, b1, acc1);
    }
  }
  const int n0 = nt0 * 16 + (lane & 15);
  const int mb = (lane & 16) ? 8 : 0;
  const float bv0 = HAS_BIAS ? bias[n0] : 0.0f;
  const float bv1 = (HAS_BIAS && has2) ? bias[n0 + 16] : 0.0f;
#pragma unroll
  for (int r = 0; r < 8; ++r) {
    float v0 = acc0[r] + bv0;
    if (RELU) v0 = fmaxf(v0, 0.0f);
    C[(m0 + mb + r) * (size_t)ldc + n0] = v0;
    if (has2) {
      float v1 = acc1[r] + bv1;
      if (RELU) v1 = fmaxf(v1, 0.0f);
      C[(m0 + mb + r) * (size_t)ldc + n0 + 16] = v1;
    }
  }
}

// ---------------------------------------------------------------------------
// GCN scatter machinery
// ---------------------------------------------------------------------------
__global__ void deg_init_kernel(float* deg, int n) {
  int i = blockIdx.x * blockDim.x + threadIdx.x;
  if (i < n) deg[i] = 1.0f;                    // self-loop
}
__global__ void deg_count_kernel(const int* __restrict__ ei, float* deg, int E) {
  int e = blockIdx.x * blockDim.x + threadIdx.x;
  if (e < E) atomicAdd(&deg[ei[E + e]], 1.0f);
}
__global__ void deg_finish_kernel(float* deg, int n) {   // deg -> dinv
  int i = blockIdx.x * blockDim.x + threadIdx.x;
  if (i < n) deg[i] = rsqrtf(deg[i]);
}
// AGG = H * dinv^2 (self-loop message)
__global__ void agg_init_kernel(const float* __restrict__ H,
                                const float* __restrict__ dinv,
                                float* __restrict__ AGG, int total, int F) {
  int i = blockIdx.x * blockDim.x + threadIdx.x;
  if (i >= total) return;
  float d = dinv[i / F];
  AGG[i] = H[i] * d * d;
}
// AGG[dst] += H[src] * dinv[src]*dinv[dst]  (one block per edge)
__global__ __launch_bounds__(256) void agg_edges_kernel(
    const int* __restrict__ ei, const float* __restrict__ H,
    const float* __restrict__ dinv, float* __restrict__ AGG, int E, int F) {
  int e = blockIdx.x;
  int s = ei[e], d = ei[E + e];
  float w = dinv[s] * dinv[d];
  const float* hs = H + (size_t)s * F;
  float* od = AGG + (size_t)d * F;
  for (int c = threadIdx.x; c < F; c += 256) atomicAdd(&od[c], hs[c] * w);
}
// X = relu(AGG + bias), emitted as bf16 for the next WMMA layer
__global__ void finalize_kernel(const float* __restrict__ AGG,
                                const float* __restrict__ bias,
                                __bf16* __restrict__ Xb, int total, int F) {
  int i = blockIdx.x * blockDim.x + threadIdx.x;
  if (i >= total) return;
  Xb[i] = (__bf16)fmaxf(AGG[i] + bias[i % F], 0.0f);
}
// final head: out[n] = X3[n,:] . fc2_w + fc2_b
__global__ void fc2_kernel(const float* __restrict__ X3,
                           const float* __restrict__ w,
                           const float* __restrict__ b,
                           float* __restrict__ out, int n) {
  int i = blockIdx.x * blockDim.x + threadIdx.x;
  if (i >= n) return;
  float s = b[0];
  const float* x = X3 + (size_t)i * F3;
#pragma unroll 8
  for (int k = 0; k < F3; ++k) s += x[k] * w[k];
  out[i] = s;
}

// ---------------------------------------------------------------------------
// Host launcher
// ---------------------------------------------------------------------------
static inline size_t align256(size_t x) { return (x + 255) & ~(size_t)255; }

extern "C" void kernel_launch(void* const* d_in, const int* in_sizes, int n_in,
                              void* d_out, int out_size, void* d_ws, size_t ws_size,
                              hipStream_t stream) {
  const float* scalar_in = (const float*)d_in[0];
  const float* fv        = (const float*)d_in[1];
  const int*   ei        = (const int*)d_in[2];
  const float* in_w      = (const float*)d_in[3];
  const float* in_b      = (const float*)d_in[4];
  const float* out_w     = (const float*)d_in[5];
  const float* out_b     = (const float*)d_in[6];
  const float* ln_g      = (const float*)d_in[7];
  const float* ln_b      = (const float*)d_in[8];
  const float* g1_w      = (const float*)d_in[9];
  const float* g1_b      = (const float*)d_in[10];
  const float* g2_w      = (const float*)d_in[11];
  const float* g2_b      = (const float*)d_in[12];
  const float* fc1_w     = (const float*)d_in[13];
  const float* fc1_b     = (const float*)d_in[14];
  const float* fc2_w     = (const float*)d_in[15];
  const float* fc2_b     = (const float*)d_in[16];
  float* out = (float*)d_out;

  const int N = in_sizes[0];
  const int E = in_sizes[2] / 2;

  // workspace carve-out
  char* ws = (char*)d_ws;
  size_t off = 0;
  auto carve = [&](size_t bytes) { char* p = ws + off; off += align256(bytes); return p; };
  __bf16* wq_bf   = (__bf16*)carve((size_t)3 * DMODEL * DMODEL * 2);
  __bf16* wo_bf   = (__bf16*)carve((size_t)DMODEL * DMODEL * 2);
  __bf16* wg1t    = (__bf16*)carve((size_t)F1 * KPAD * 2);
  __bf16* wg2t    = (__bf16*)carve((size_t)F2 * F1 * 2);
  __bf16* wfc1    = (__bf16*)carve((size_t)F3 * F2 * 2);
  __bf16* nf      = (__bf16*)carve((size_t)N * KPAD * 2);
  float*  H1      = (float*)carve((size_t)N * F1 * 4);
  float*  AGG1    = (float*)carve((size_t)N * F1 * 4);
  __bf16* X1b     = (__bf16*)carve((size_t)N * F1 * 2);
  float*  H2      = (float*)carve((size_t)N * F2 * 4);
  float*  AGG2    = (float*)carve((size_t)N * F2 * 4);
  __bf16* X2b     = (__bf16*)carve((size_t)N * F2 * 2);
  float*  X3      = (float*)carve((size_t)N * F3 * 4);
  float*  dinv    = (float*)carve((size_t)N * 4);
  (void)ws_size;

  const int T = 256;
  auto blks = [&](size_t n) { return (unsigned)((n + T - 1) / T); };

  // --- weight prep ---
  cvt_bf16_kernel<<<blks((size_t)3 * DMODEL * DMODEL), T, 0, stream>>>(in_w, wq_bf, 3 * DMODEL * DMODEL);
  cvt_bf16_kernel<<<blks((size_t)DMODEL * DMODEL), T, 0, stream>>>(out_w, wo_bf, DMODEL * DMODEL);
  cvt_bf16_kernel<<<blks((size_t)F3 * F2), T, 0, stream>>>(fc1_w, wfc1, F3 * F2);
  transpose_cvt_kernel<<<blks((size_t)F1 * KPAD), T, 0, stream>>>(g1_w, wg1t, DMODEL + 1, F1, KPAD);
  transpose_cvt_kernel<<<blks((size_t)F2 * F1), T, 0, stream>>>(g2_w, wg2t, F1, F2, F1);

  // --- fused attention + layernorm + concat (1 block per node) ---
  const int SMEM = (16 * DMODEL + 16 * 3 * DMODEL + 16 * 32) * 2 + (DMODEL + 512) * 4;
  (void)hipFuncSetAttribute((const void*)attn_kernel,
                            hipFuncAttributeMaxDynamicSharedMemorySize, SMEM);
  attn_kernel<<<N, 256, SMEM, stream>>>(fv, scalar_in, wq_bf, in_b, wo_bf, out_b,
                                        ln_g, ln_b, nf);

  // --- GCN normalization ---
  deg_init_kernel<<<blks(N), T, 0, stream>>>(dinv, N);
  deg_count_kernel<<<blks(E), T, 0, stream>>>(ei, dinv, E);
  deg_finish_kernel<<<blks(N), T, 0, stream>>>(dinv, N);

  const int Mtiles = N / 16;  // 625

  // --- GCN layer 1: H1 = nf @ W1 (K=800, 25 steps; 32 n-tiles = 8/block) ---
  gemm_kernel<false, false><<<dim3(Mtiles, F1 / 128), 128, 0, stream>>>(
      nf, KPAD, wg1t, KPAD, nullptr, H1, F1, F1 / 16, KPAD / 32);
  agg_init_kernel<<<blks((size_t)N * F1), T, 0, stream>>>(H1, dinv, AGG1, N * F1, F1);
  agg_edges_kernel<<<E, 256, 0, stream>>>(ei, H1, dinv, AGG1, E, F1);
  finalize_kernel<<<blks((size_t)N * F1), T, 0, stream>>>(AGG1, g1_b, X1b, N * F1, F1);

  // --- GCN layer 2: H2 = X1 @ W2 (K=512, 16 steps; 16 n-tiles = 8/block) ---
  gemm_kernel<false, false><<<dim3(Mtiles, F2 / 128), 128, 0, stream>>>(
      X1b, F1, wg2t, F1, nullptr, H2, F2, F2 / 16, F1 / 32);
  agg_init_kernel<<<blks((size_t)N * F2), T, 0, stream>>>(H2, dinv, AGG2, N * F2, F2);
  agg_edges_kernel<<<E, 256, 0, stream>>>(ei, H2, dinv, AGG2, E, F2);
  finalize_kernel<<<blks((size_t)N * F2), T, 0, stream>>>(AGG2, g2_b, X2b, N * F2, F2);

  // --- FC head ---
  gemm_kernel<true, true><<<dim3(Mtiles, 1), 128, 0, stream>>>(
      X2b, F2, wfc1, F2, fc1_b, X3, F3, F3 / 16, F2 / 32);
  fc2_kernel<<<blks(N), T, 0, stream>>>(X3, fc2_w, fc2_b, out, N);
}
